// RegionProposalNetwork_30262339568124
// MI455X (gfx1250) — matrix operand order repaired and checked by reading, not compiled, not run on hardware
//
#include <hip/hip_runtime.h>
#include <hip/hip_bf16.h>

typedef _Float16 v16h __attribute__((ext_vector_type(16)));
typedef _Float16 v8h  __attribute__((ext_vector_type(8)));
typedef float    v8f  __attribute__((ext_vector_type(8)));
typedef int      v4i_ __attribute__((vector_size(4 * sizeof(int))));

#define NIMG   4
#define CIN    512
#define CMID   512
#define HH     64
#define WW     64
#define HP     66              // padded spatial
#define NPIX   (NIMG * HH * WW)        // 16384
#define NANCH  (HH * WW * 9)           // 36864 per image
#define PRE_N  3000
#define POST_N 300
#define KTOT   (9 * CIN)               // 4608
#define CTILE  256                     // conv: out-channels per block
#define NSTEPS (9 * (CIN / 32))        // 144 K-steps of 32

// ---------------------------------------------------------------------------
// CDNA5 async global->LDS staging (guarded: plain b128 copy fallback)
// ---------------------------------------------------------------------------
#if defined(__has_builtin)
#if __has_builtin(__builtin_amdgcn_global_load_async_to_lds_b128)
#define HAVE_ASYNC 1
#endif
#endif

#ifdef HAVE_ASYNC
__device__ __forceinline__ void stage_b128(void* gsrc, void* ldst) {
    __builtin_amdgcn_global_load_async_to_lds_b128(
        (__attribute__((address_space(1))) v4i_*)gsrc,
        (__attribute__((address_space(3))) v4i_*)ldst,
        0, 0);
}
#if __has_builtin(__builtin_amdgcn_s_wait_asynccnt)
#define ASYNC_WAIT() __builtin_amdgcn_s_wait_asynccnt(0)
#else
#define ASYNC_WAIT() asm volatile("s_wait_asynccnt 0" ::: "memory")
#endif
#else
__device__ __forceinline__ void stage_b128(void* gsrc, void* ldst) {
    *(ulonglong2*)ldst = *(const ulonglong2*)gsrc;   // load + ds_store fallback
}
#define ASYNC_WAIT()
#endif

__device__ __forceinline__ v16h frag_cat(v8h lo, v8h hi) {
    return __builtin_shufflevector(lo, hi, 0, 1, 2, 3, 4, 5, 6, 7,
                                   8, 9, 10, 11, 12, 13, 14, 15);
}

// ---------------------------------------------------------------------------
// prep kernels
// ---------------------------------------------------------------------------
__global__ void zero_f16_kernel(_Float16* __restrict__ p, int n) {
    int i = blockIdx.x * blockDim.x + threadIdx.x;
    if (i < n) p[i] = (_Float16)0.0f;
}

// xt[n][y+1][x+1][c] = x[n][c][y][x]   (padded NHWC, halo stays zero)
__global__ void pack_x_kernel(const float* __restrict__ x,
                              _Float16* __restrict__ xt) {
    int i = blockIdx.x * blockDim.x + threadIdx.x;
    if (i >= NIMG * CIN * HH * WW) return;
    int n  = i / (CIN * HH * WW);
    int c  = (i / (HH * WW)) % CIN;
    int yy = (i / WW) % HH;
    int xx = i % WW;
    xt[(((size_t)(n * HP + yy + 1) * HP) + xx + 1) * CIN + c] = (_Float16)x[i];
}

// Bt[m][r*512 + c] = w_conv[m][c][r]
__global__ void transpose_w_kernel(const float* __restrict__ w,
                                   _Float16* __restrict__ Bt) {
    int i = blockIdx.x * blockDim.x + threadIdx.x;     // over 512*4608
    if (i >= CMID * KTOT) return;
    int m = i / KTOT;
    int k = i % KTOT;
    int r = k >> 9;
    int c = k & 511;
    Bt[i] = (_Float16)w[((size_t)m * CIN + c) * 9 + r];
}

// pack loc(36) + score(18) weights into a padded [64][512] f16 matrix + bias
__global__ void pack_heads_kernel(const float* __restrict__ w_loc,
                                  const float* __restrict__ b_loc,
                                  const float* __restrict__ w_score,
                                  const float* __restrict__ b_score,
                                  _Float16* __restrict__ Wh,
                                  float* __restrict__ bh) {
    int i = blockIdx.x * blockDim.x + threadIdx.x;     // 64*512
    if (i >= 64 * CIN) return;
    int m = i >> 9, c = i & 511;
    float v = 0.0f;
    if (m < 36)      v = w_loc[(size_t)m * CIN + c];
    else if (m < 54) v = w_score[(size_t)(m - 36) * CIN + c];
    Wh[i] = (_Float16)v;
    if (c == 0) {
        float b = 0.0f;
        if (m < 36)      b = b_loc[m];
        else if (m < 54) b = b_score[m - 36];
        bh[m] = b;
    }
}

// ---------------------------------------------------------------------------
// 3x3 conv + bias + ReLU: implicit GEMM, WMMA f16->f32, double-buffered
// async LDS staging. block 256 (8 waves), tile 64 pixels x 256 channels.
// grid: (NPIX/64, CMID/CTILE)
// ---------------------------------------------------------------------------
__global__ __launch_bounds__(256) void conv3x3_wmma_kernel(
    const _Float16* __restrict__ xt,    // [4][66][66][512]
    const _Float16* __restrict__ Bt,    // [512][4608]
    const float*    __restrict__ bias,  // [512]
    _Float16*       __restrict__ feat)  // [NPIX][512]
{
    __shared__ __align__(64) _Float16 Alds[2][64 * 32];     //  4 KB x2
    __shared__ __align__(64) _Float16 Blds[2][CTILE * 32];  // 16 KB x2

    const int tid  = threadIdx.x;
    const int lane = tid & 31;
    const int wave = tid >> 5;
    const int wp   = wave >> 2;          // 0..1 : 32-pixel half
    const int wq   = wave & 3;           // 0..3 : 64-channel quarter
    const int g    = lane >> 4;
    const int lr   = lane & 15;

    const int ptile = blockIdx.x;        // (n, y) pixel row
    const int n     = ptile >> 6;
    const int y     = ptile & 63;
    const int cbase = blockIdx.y * CTILE;

    const int ax   = tid >> 2;           // staging: pixel 0..63
    const int asub = tid & 3;            // staging: 8-half chunk

    v8f acc[2][4] = {};

    auto stage = [&](int buf, int step) {
        const int r  = step >> 4;
        const int kc = (step & 15) << 5;
        const int dy = r / 3 - 1, dx = r % 3 - 1;
        const int yy = y + dy;
        // A tile: 64 pixels x 32 channels, halo already zero-padded
        {
            const _Float16* src = xt +
                ((size_t)((n * HP + yy + 1) * HP + (ax + dx + 1)) * CIN +
                 kc + asub * 8);
            stage_b128((void*)src, (void*)&Alds[buf][ax * 32 + asub * 8]);
        }
        // B tile: 256 cols x 32 k (fragment-major), 4 chunks per thread
        const size_t koff = (size_t)r * CIN + kc;
        #pragma unroll
        for (int c4 = 0; c4 < 4; ++c4) {
            int q   = tid + 256 * c4;
            int col = q >> 2, sub = q & 3;
            const _Float16* src =
                Bt + (size_t)(cbase + col) * KTOT + koff + sub * 8;
            stage_b128((void*)src, (void*)&Blds[buf][col * 32 + sub * 8]);
        }
        // L2 prefetch of upcoming weight lines
        __builtin_prefetch(
            (const void*)(Bt + (size_t)(cbase + (tid >> 2)) * KTOT + koff + 32),
            0, 1);
    };

    stage(0, 0);
    ASYNC_WAIT();
    __syncthreads();

    int cur = 0;
    for (int step = 0; step < NSTEPS; ++step) {
        if (step + 1 < NSTEPS) stage(cur ^ 1, step + 1);

        const _Float16* Ab = &Alds[cur][0];
        const _Float16* Bb = &Blds[cur][0];

        // A frags: elements e -> k = 16g + e (contiguous 32B)
        v16h af[2];
        #pragma unroll
        for (int t = 0; t < 2; ++t) {
            const int prow = wp * 32 + t * 16 + lr;
            af[t] = *(const v16h*)(Ab + prow * 32 + 16 * g);
        }
        // B frags: elements e -> k = 16g + e (contiguous 32B)
        v16h bf[4];
        #pragma unroll
        for (int t = 0; t < 4; ++t) {
            const int ncol = wq * 64 + t * 16 + lr;
            bf[t] = *(const v16h*)(Bb + ncol * 32 + g * 16);
        }
        #pragma unroll
        for (int i = 0; i < 2; ++i)
            #pragma unroll
            for (int j = 0; j < 4; ++j)
                acc[i][j] = __builtin_amdgcn_wmma_f32_16x16x32_f16(
                    false, af[i], false, bf[j], (short)0, acc[i][j],
                    false, false);

        ASYNC_WAIT();
        __syncthreads();
        cur ^= 1;
    }

    // bias + ReLU + store (C/D layout: M = vgpr + 8*half, N = lane%16)
    #pragma unroll
    for (int i = 0; i < 2; ++i) {
        #pragma unroll
        for (int j = 0; j < 4; ++j) {
            const int ncol = cbase + wq * 64 + j * 16 + lr;
            const float bv = bias[ncol];
            #pragma unroll
            for (int e = 0; e < 8; ++e) {
                const int prow = wp * 32 + i * 16 + e + 8 * g;   // = x coord
                float v = acc[i][j][e] + bv;
                v = v > 0.0f ? v : 0.0f;
                feat[(size_t)(ptile * 64 + prow) * CMID + ncol] = (_Float16)v;
            }
        }
    }
}

// ---------------------------------------------------------------------------
// 1x1 heads as WMMA GEMM: [128 pixels] x [64 padded out-ch], K = 512
// block 256 (8 waves): wave tile 32 x 32. grid: NPIX/128
// ---------------------------------------------------------------------------
__global__ __launch_bounds__(256) void heads_wmma_kernel(
    const _Float16* __restrict__ feat,   // [NPIX][512]
    const _Float16* __restrict__ Wh,     // [64][512]
    const float*    __restrict__ bh,     // [64]
    float* __restrict__ out_locs,        // [NPIX][36]
    float* __restrict__ out_scores)      // [NPIX][18]
{
    __shared__ __align__(64) _Float16 Ah[2][128 * 32];   // 8 KB x2
    __shared__ __align__(64) _Float16 Bh[2][64 * 32];    // 4 KB x2

    const int tid  = threadIdx.x;
    const int lane = tid & 31;
    const int wave = tid >> 5;
    const int wp   = wave >> 1;          // 0..3 : 32-pixel quarter
    const int wq   = wave & 1;           // 0..1 : 32-channel half
    const int g    = lane >> 4;
    const int lr   = lane & 15;
    const int pbase = blockIdx.x * 128;

    v8f acc[2][2] = {};

    auto stage = [&](int buf, int step) {
        const int kc = step << 5;
        #pragma unroll
        for (int c2 = 0; c2 < 2; ++c2) {            // A: 512 chunks
            int q = tid + 256 * c2;
            int pix = q >> 2, sub = q & 3;
            const _Float16* src =
                feat + (size_t)(pbase + pix) * CMID + kc + sub * 8;
            stage_b128((void*)src, (void*)&Ah[buf][pix * 32 + sub * 8]);
        }
        {                                            // B: 256 chunks
            int col = tid >> 2, sub = tid & 3;
            const _Float16* src = Wh + (size_t)col * CIN + kc + sub * 8;
            stage_b128((void*)src, (void*)&Bh[buf][col * 32 + sub * 8]);
        }
    };

    stage(0, 0);
    ASYNC_WAIT();
    __syncthreads();

    int cur = 0;
    for (int step = 0; step < CIN / 32; ++step) {
        if (step + 1 < CIN / 32) stage(cur ^ 1, step + 1);

        const _Float16* Ab = &Ah[cur][0];
        const _Float16* Bb = &Bh[cur][0];
        v16h af[2], bf[2];
        #pragma unroll
        for (int t = 0; t < 2; ++t) {
            const int prow = wp * 32 + t * 16 + lr;
            af[t] = *(const v16h*)(Ab + prow * 32 + 16 * g);
        }
        #pragma unroll
        for (int t = 0; t < 2; ++t) {
            const int ncol = wq * 32 + t * 16 + lr;
            bf[t] = *(const v16h*)(Bb + ncol * 32 + g * 16);
        }
        #pragma unroll
        for (int i = 0; i < 2; ++i)
            #pragma unroll
            for (int j = 0; j < 2; ++j)
                acc[i][j] = __builtin_amdgcn_wmma_f32_16x16x32_f16(
                    false, af[i], false, bf[j], (short)0, acc[i][j],
                    false, false);

        ASYNC_WAIT();
        __syncthreads();
        cur ^= 1;
    }

    #pragma unroll
    for (int i = 0; i < 2; ++i) {
        #pragma unroll
        for (int j = 0; j < 2; ++j) {
            const int o = wq * 32 + j * 16 + lr;
            const float bv = bh[o];
            #pragma unroll
            for (int e = 0; e < 8; ++e) {
                const int p = pbase + wp * 32 + i * 16 + e + 8 * g;
                float v = acc[i][j][e] + bv;
                if (o < 36)      out_locs[(size_t)p * 36 + o] = v;
                else if (o < 54) out_scores[(size_t)p * 18 + (o - 36)] = v;
            }
        }
    }
}

// ---------------------------------------------------------------------------
// softmax fg, anchor decode, top-k + NMS (serial semantics match reference)
// ---------------------------------------------------------------------------
__global__ void softmax_fg_kernel(const float* __restrict__ scores,
                                  float* __restrict__ fg) {
    int i = blockIdx.x * blockDim.x + threadIdx.x;
    if (i >= NIMG * NANCH) return;
    float s0 = scores[(size_t)i * 2 + 0];
    float s1 = scores[(size_t)i * 2 + 1];
    float m  = fmaxf(s0, s1);
    float e0 = __expf(s0 - m), e1 = __expf(s1 - m);
    fg[i] = e1 / (e0 + e1);
}

__device__ inline void anchor_for(int a, int x, int y,
                                  float& x0, float& y0, float& x1, float& y1) {
    const float ratios[3] = {0.5f, 1.0f, 2.0f};
    const float scales[3] = {8.0f, 16.0f, 32.0f};
    int ri = a / 3, si = a % 3;
    float h = 16.0f * scales[si] * __builtin_sqrtf(ratios[ri]);
    float w = 16.0f * scales[si] * __builtin_sqrtf(1.0f / ratios[ri]);
    float px = 8.0f + 16.0f * x, py = 8.0f + 16.0f * y;
    x0 = px - 0.5f * w; y0 = py - 0.5f * h;
    x1 = px + 0.5f * w; y1 = py + 0.5f * h;
}

__global__ void decode_kernel(const float* __restrict__ locs,
                              const float* __restrict__ fg,
                              float* __restrict__ boxes,
                              float* __restrict__ scores_ws,
                              float* __restrict__ out_anchor,
                              const int* __restrict__ img_hp,
                              const int* __restrict__ img_wp) {
    int i = blockIdx.x * blockDim.x + threadIdx.x;
    if (i >= NIMG * NANCH) return;
    const float img_h = (float)img_hp[0];
    const float img_w = (float)img_wp[0];
    int n  = i / NANCH;
    int ia = i % NANCH;
    int a  = ia % 9;
    int yx = ia / 9;
    int y  = yx / WW, x = yx % WW;

    float ax0, ay0, ax1, ay1;
    anchor_for(a, x, y, ax0, ay0, ax1, ay1);
    if (n == 0) {
        out_anchor[(size_t)ia * 4 + 0] = ax0;
        out_anchor[(size_t)ia * 4 + 1] = ay0;
        out_anchor[(size_t)ia * 4 + 2] = ax1;
        out_anchor[(size_t)ia * 4 + 3] = ay1;
    }
    float w  = ax1 - ax0, h = ay1 - ay0;
    float cx = ax0 + 0.5f * w, cy = ay0 + 0.5f * h;
    const float* l = locs + (size_t)i * 4;
    float ncx = l[0] * w + cx, ncy = l[1] * h + cy;
    float nw  = __expf(l[2]) * w, nh = __expf(l[3]) * h;
    float x0 = fminf(fmaxf(ncx - 0.5f * nw, 0.0f), img_w);
    float y0 = fminf(fmaxf(ncy - 0.5f * nh, 0.0f), img_h);
    float x1 = fminf(fmaxf(ncx + 0.5f * nw, 0.0f), img_w);
    float y1 = fminf(fmaxf(ncy + 0.5f * nh, 0.0f), img_h);
    boxes[(size_t)i * 4 + 0] = x0;
    boxes[(size_t)i * 4 + 1] = y0;
    boxes[(size_t)i * 4 + 2] = x1;
    boxes[(size_t)i * 4 + 3] = y1;
    bool valid = ((x1 - x0) >= 16.0f) && ((y1 - y0) >= 16.0f);
    scores_ws[i] = valid ? fg[i] : -1e9f;
}

__global__ __launch_bounds__(1024) void topk_nms_kernel(
    const float* __restrict__ boxes_all,
    float* __restrict__ scores,
    float* __restrict__ sorted_boxes,
    int*   __restrict__ keep,
    float* __restrict__ out_rois) {
    __shared__ float smax[1024];
    __shared__ int   simax[1024];

    const int n   = blockIdx.x;
    const int tid = threadIdx.x;
    const float* bx = boxes_all + (size_t)n * NANCH * 4;
    float* sc = scores       + (size_t)n * NANCH;
    float* sb = sorted_boxes + (size_t)n * PRE_N * 4;
    int*   kp = keep         + (size_t)n * PRE_N;

    for (int it = 0; it < PRE_N; ++it) {
        float best = -3.0e38f; int bi = 0;
        for (int j = tid; j < NANCH; j += 1024) {
            float v = sc[j];
            if (v > best) { best = v; bi = j; }
        }
        smax[tid] = best; simax[tid] = bi;
        __syncthreads();
        for (int s = 512; s > 0; s >>= 1) {
            if (tid < s && smax[tid + s] > smax[tid]) {
                smax[tid] = smax[tid + s]; simax[tid] = simax[tid + s];
            }
            __syncthreads();
        }
        if (tid == 0) {
            int w = simax[0];
            sc[w] = -3.4e38f;
            sb[it * 4 + 0] = bx[(size_t)w * 4 + 0];
            sb[it * 4 + 1] = bx[(size_t)w * 4 + 1];
            sb[it * 4 + 2] = bx[(size_t)w * 4 + 2];
            sb[it * 4 + 3] = bx[(size_t)w * 4 + 3];
        }
        __syncthreads();
    }

    for (int i = 0; i < PRE_N; ++i) {
        float ix0 = sb[i * 4 + 0], iy0 = sb[i * 4 + 1];
        float ix1 = sb[i * 4 + 2], iy1 = sb[i * 4 + 3];
        float iarea = (ix1 - ix0) * (iy1 - iy0);
        int any = 0;
        for (int j = tid; j < i; j += 1024) {
            if (kp[j]) {
                float jx0 = sb[j * 4 + 0], jy0 = sb[j * 4 + 1];
                float jx1 = sb[j * 4 + 2], jy1 = sb[j * 4 + 3];
                float xx0 = fmaxf(ix0, jx0), yy0 = fmaxf(iy0, jy0);
                float xx1 = fminf(ix1, jx1), yy1 = fminf(iy1, jy1);
                float inter = fmaxf(xx1 - xx0, 0.0f) * fmaxf(yy1 - yy0, 0.0f);
                float jarea = (jx1 - jx0) * (jy1 - jy0);
                float iou = inter / (iarea + jarea - inter + 1e-9f);
                if (iou > 0.7f) any = 1;
            }
        }
        simax[tid] = any;
        __syncthreads();
        for (int s = 512; s > 0; s >>= 1) {
            if (tid < s) simax[tid] |= simax[tid + s];
            __syncthreads();
        }
        if (tid == 0) kp[i] = simax[0] ? 0 : 1;
        __syncthreads();
    }

    if (tid == 0) {
        int cnt = 0;
        for (int i = 0; i < PRE_N && cnt < POST_N; ++i) {
            if (kp[i]) {
                float* o = out_rois + ((size_t)n * POST_N + cnt) * 4;
                o[0] = sb[i * 4 + 0]; o[1] = sb[i * 4 + 1];
                o[2] = sb[i * 4 + 2]; o[3] = sb[i * 4 + 3];
                ++cnt;
            }
        }
        for (; cnt < POST_N; ++cnt) {
            float* o = out_rois + ((size_t)n * POST_N + cnt) * 4;
            o[0] = sb[0]; o[1] = sb[1]; o[2] = sb[2]; o[3] = sb[3];
        }
    }
}

__global__ void roi_indices_kernel(float* __restrict__ out_idx) {
    int i = blockIdx.x * blockDim.x + threadIdx.x;
    if (i < NIMG * POST_N) out_idx[i] = (float)(i / POST_N);
}

// ---------------------------------------------------------------------------
extern "C" void kernel_launch(void* const* d_in, const int* in_sizes, int n_in,
                              void* d_out, int out_size, void* d_ws, size_t ws_size,
                              hipStream_t stream) {
    const float* x       = (const float*)d_in[0];
    const float* w_conv  = (const float*)d_in[1];
    const float* b_conv  = (const float*)d_in[2];
    const float* w_score = (const float*)d_in[3];
    const float* b_score = (const float*)d_in[4];
    const float* w_loc   = (const float*)d_in[5];
    const float* b_loc   = (const float*)d_in[6];
    const int*   img_h   = (const int*)d_in[7];
    const int*   img_w   = (const int*)d_in[8];

    float* out_locs   = (float*)d_out;
    float* out_scores = out_locs + (size_t)NIMG * NANCH * 4;
    float* out_rois   = out_scores + (size_t)NIMG * NANCH * 2;
    float* out_ridx   = out_rois + (size_t)NIMG * POST_N * 4;
    float* out_anchor = out_ridx + (size_t)NIMG * POST_N;

    char* ws = (char*)d_ws;
    auto take = [&](size_t bytes) {
        char* p = ws;
        ws += (bytes + 255) & ~(size_t)255;
        return p;
    };
    _Float16* xt        = (_Float16*)take((size_t)NIMG * HP * HP * CIN * 2);
    _Float16* Bt        = (_Float16*)take((size_t)CMID * KTOT * 2);
    _Float16* feat      = (_Float16*)take((size_t)NPIX * CMID * 2);
    _Float16* Wh        = (_Float16*)take((size_t)64 * CIN * 2);
    float*    bh        = (float*)take((size_t)64 * 4);
    float*    fg        = (float*)take((size_t)NIMG * NANCH * 4);
    float*    boxes     = (float*)take((size_t)NIMG * NANCH * 4 * 4);
    float*    scores_ws = (float*)take((size_t)NIMG * NANCH * 4);
    float*    sortedb   = (float*)take((size_t)NIMG * PRE_N * 4 * 4);
    int*      keep      = (int*)take((size_t)NIMG * PRE_N * 4);

    {   // zero padded activation tensor (halo), then pack interior
        int n = NIMG * HP * HP * CIN;
        zero_f16_kernel<<<(n + 255) / 256, 256, 0, stream>>>(xt, n);
    }
    {
        int n = NIMG * CIN * HH * WW;
        pack_x_kernel<<<(n + 255) / 256, 256, 0, stream>>>(x, xt);
    }
    {
        int n = CMID * KTOT;
        transpose_w_kernel<<<(n + 255) / 256, 256, 0, stream>>>(w_conv, Bt);
    }
    {
        int n = 64 * CIN;
        pack_heads_kernel<<<(n + 255) / 256, 256, 0, stream>>>(
            w_loc, b_loc, w_score, b_score, Wh, bh);
    }

    {   // 3x3 conv + ReLU
        dim3 grid(NPIX / 64, CMID / CTILE);
        conv3x3_wmma_kernel<<<grid, 256, 0, stream>>>(xt, Bt, b_conv, feat);
    }

    // 1x1 heads
    heads_wmma_kernel<<<NPIX / 128, 256, 0, stream>>>(feat, Wh, bh,
                                                      out_locs, out_scores);

    {
        int n = NIMG * NANCH;
        softmax_fg_kernel<<<(n + 255) / 256, 256, 0, stream>>>(out_scores, fg);
        decode_kernel<<<(n + 255) / 256, 256, 0, stream>>>(
            out_locs, fg, boxes, scores_ws, out_anchor, img_h, img_w);
    }

    topk_nms_kernel<<<NIMG, 1024, 0, stream>>>(boxes, scores_ws, sortedb, keep,
                                               out_rois);
    roi_indices_kernel<<<(NIMG * POST_N + 255) / 256, 256, 0, stream>>>(out_ridx);
}
